// WubuBlock_76201309766273
// MI455X (gfx1250) — compile-verified, matrix operand order checked by reading, not run on hardware
//
#include <hip/hip_runtime.h>
#include <hip/hip_bf16.h>
#include <stddef.h>
#include <stdint.h>

#define EPS      1e-7f
#define ONE_EPS  (1.0f - 1e-7f)
#define LN_EPS   1e-6f

typedef __attribute__((ext_vector_type(16))) __bf16 v16bf;
typedef __attribute__((ext_vector_type(8)))  float  v8f;
typedef __attribute__((ext_vector_type(4)))  unsigned int v4u;
typedef __attribute__((ext_vector_type(4)))  int v4i;

typedef __attribute__((address_space(1))) v4i g_v4i;   // global
typedef __attribute__((address_space(3))) v4i l_v4i;   // LDS

union Frag16 { v16bf v; v4u u[2]; };

// ---------- CDNA5 async global->LDS (ASYNCcnt path) ----------

__device__ inline void async_ld_b128(void* lds, const void* g) {
#if __has_builtin(__builtin_amdgcn_global_load_async_to_lds_b128)
    __builtin_amdgcn_global_load_async_to_lds_b128(
        (g_v4i*)(uintptr_t)g,      // global source (AS1), 64-bit flat addr == global addr
        (l_v4i*)(uintptr_t)lds,    // LDS dest (AS3), low 32 bits of flat addr == LDS offset
        0, 0);
#else
    unsigned loff = (unsigned)(uintptr_t)lds;
    asm volatile("global_load_async_to_lds_b128 %0, %1, off"
                 :: "v"(loff), "v"(g) : "memory");
#endif
}

#if __has_builtin(__builtin_amdgcn_s_wait_asynccnt)
#define WAIT_ASYNC(n) __builtin_amdgcn_s_wait_asynccnt(n)
#else
#define WAIT_ASYNC(n) asm volatile("s_wait_asynccnt %0" :: "n"(n) : "memory")
#endif

// ---------- small helpers ----------

__device__ inline unsigned short f2bf(float f) {
    unsigned int u = __float_as_uint(f);
    unsigned int r = u + 0x7FFFu + ((u >> 16) & 1u);   // round-to-nearest-even
    return (unsigned short)(r >> 16);
}

__device__ inline float gelu_tanh(float x) {
    float x3 = x * x * x;
    return 0.5f * x * (1.f + tanhf(0.7978845608f * (x + 0.044715f * x3)));
}

// block-wide sum, 256 threads (8 waves of 32)
__device__ inline float bsum(float v, float* red) {
    #pragma unroll
    for (int m = 16; m >= 1; m >>= 1) v += __shfl_xor(v, m, 32);
    int w = threadIdx.x >> 5;
    if ((threadIdx.x & 31) == 0) red[w] = v;
    __syncthreads();
    if (threadIdx.x == 0) {
        float s = 0.f;
        #pragma unroll
        for (int i = 0; i < 8; ++i) s += red[i];
        red[0] = s;
    }
    __syncthreads();
    float s = red[0];
    __syncthreads();
    return s;
}

// block-wide sum, 64 threads (2 waves)
__device__ inline float blk64_sum(float v, float* red) {
    #pragma unroll
    for (int m = 16; m >= 1; m >>= 1) v += __shfl_xor(v, m, 32);
    if ((threadIdx.x & 31) == 0) red[threadIdx.x >> 5] = v;
    __syncthreads();
    float s = red[0] + red[1];
    __syncthreads();
    return s;
}

// Load a 16x32 bf16 WMMA A/B fragment from LDS (row-major, `stride` bf16 per row,
// sub-tile row base r0, K-offset kof). Matches CDNA5 16-bit A-matrix layout:
// lane half=0 holds K = {0..7, 16..23}, half=1 holds K = {8..15, 24..31}.
__device__ inline v16bf ldsfrag(const unsigned short* lds, int r0, int stride,
                                int kof, int lane) {
    int row  = r0 + (lane & 15);
    int half = lane >> 4;
    const unsigned short* p = lds + row * stride + kof + half * 8;
    Frag16 f;
    f.u[0] = *(const v4u*)(p);
    f.u[1] = *(const v4u*)(p + 16);
    return f.v;
}

__device__ inline v8f wmma_bf16(v16bf a, v16bf b, v8f c) {
    return __builtin_amdgcn_wmma_f32_16x16x32_bf16(
        false, a, false, b, (short)0, c, false, false);
}

// ---------- hyperbolic row ops (row of 1024, 256 threads, 4 elems/thread) ----------

__device__ inline void expmap_row(const float vv[4], float c, float sc,
                                  float out[4], float* red) {
    float s = 0.f;
    #pragma unroll
    for (int j = 0; j < 4; ++j) s += vv[j] * vv[j];
    float n = sqrtf(bsum(s, red));
    float safe = fmaxf(n, EPS);
    float mag = tanhf(sc * safe) / sc;
    float pn = (n < EPS) ? 0.f : mag;
    float ps = (pn >= 1.f) ? ONE_EPS / pn : 1.f;
    float coef = (n < EPS) ? 0.f : (mag / safe) * ps;
    #pragma unroll
    for (int j = 0; j < 4; ++j) out[j] = coef * vv[j];
}

__device__ inline void mobius_row(const float xv[4], const float yv[4], float c,
                                  float out[4], float* red) {
    float sx = 0.f, sy = 0.f, sxy = 0.f;
    #pragma unroll
    for (int j = 0; j < 4; ++j) { sx += xv[j]*xv[j]; sy += yv[j]*yv[j]; sxy += xv[j]*yv[j]; }
    float x2 = bsum(sx, red);
    float y2 = bsum(sy, red);
    float xy = bsum(sxy, red);
    float A = 1.f + 2.f*c*xy + c*y2;
    float B = 1.f - c*x2;
    float den = fmaxf(1.f + 2.f*c*xy + c*c*x2*y2, EPS);
    float vloc[4]; float sv = 0.f;
    #pragma unroll
    for (int j = 0; j < 4; ++j) { vloc[j] = (A*xv[j] + B*yv[j]) / den; sv += vloc[j]*vloc[j]; }
    float vn = sqrtf(bsum(sv, red));
    float ps = (vn >= 1.f) ? ONE_EPS / fmaxf(vn, EPS) : 1.f;
    #pragma unroll
    for (int j = 0; j < 4; ++j) out[j] = vloc[j] * ps;
}

// x_tan = logmap0(expmap0(LN(logmap0(x))))  -> bf16 row
__device__ inline void prenorm_row(const float xv[4], float c, float sc,
                                   const float* __restrict__ scale,
                                   const float* __restrict__ bias,
                                   unsigned short* __restrict__ outRow,
                                   float* red) {
    int tid = threadIdx.x;
    float s1 = 0.f;
    #pragma unroll
    for (int j = 0; j < 4; ++j) s1 += xv[j]*xv[j];
    float n1 = sqrtf(bsum(s1, red));
    float coef = (n1 < EPS) ? 0.f : atanhf(fminf(n1, ONE_EPS)) / sc / fmaxf(n1, EPS);
    float u[4];
    #pragma unroll
    for (int j = 0; j < 4; ++j) u[j] = coef * xv[j];
    float mu = bsum(u[0]+u[1]+u[2]+u[3], red) * (1.f/1024.f);
    float sv = 0.f;
    #pragma unroll
    for (int j = 0; j < 4; ++j) { float d = u[j]-mu; sv += d*d; }
    float var = bsum(sv, red) * (1.f/1024.f);
    float rs = rsqrtf(var + LN_EPS);
    float ln[4]; float s2 = 0.f;
    #pragma unroll
    for (int j = 0; j < 4; ++j) {
        int i = tid + j*256;
        ln[j] = (u[j]-mu)*rs*scale[i] + bias[i];
        s2 += ln[j]*ln[j];
    }
    float n2 = sqrtf(bsum(s2, red));
    float safe2 = fmaxf(n2, EPS);
    float mag = tanhf(sc * safe2) / sc;
    float pn = (n2 < EPS) ? 0.f : mag;
    float ps  = (pn >= 1.f) ? ONE_EPS / pn : 1.f;
    float pnp = (pn >= 1.f) ? ONE_EPS : pn;
    float ycoef = (n2 < EPS) ? 0.f : (mag / safe2) * ps;
    float c2 = (pnp < EPS) ? 0.f : atanhf(fminf(pnp, ONE_EPS)) / sc / fmaxf(pnp, EPS);
    float cc = c2 * ycoef;
    #pragma unroll
    for (int j = 0; j < 4; ++j) {
        int i = tid + j*256;
        outRow[i] = f2bf(cc * ln[j]);
    }
}

// ---------- kernels ----------

// fp32 [K][N] -> bf16 [N][K]  (transpose for WMMA B operand)
__global__ __launch_bounds__(256)
void wconv_kernel(const float* __restrict__ w, unsigned short* __restrict__ wt,
                  int K, int N) {
    size_t total = (size_t)K * N;
    for (size_t idx = (size_t)blockIdx.x*256 + threadIdx.x; idx < total;
         idx += (size_t)gridDim.x * 256) {
        size_t k = idx / N, n = idx % N;
        wt[n * (size_t)K + k] = f2bf(w[idx]);
    }
}

__global__ __launch_bounds__(256)
void prenorm1_kernel(const float* __restrict__ x, const float* __restrict__ cg,
                     const float* __restrict__ s, const float* __restrict__ b,
                     unsigned short* __restrict__ xtan) {
    __shared__ float red[8];
    int t = blockIdx.x, tid = threadIdx.x;
    float c = cg[0], sc = fmaxf(sqrtf(c), EPS);
    float xv[4];
    #pragma unroll
    for (int j = 0; j < 4; ++j) xv[j] = x[(size_t)t*1024 + tid + j*256];
    prenorm_row(xv, c, sc, s, b, xtan + (size_t)t*1024, red);
}

// Generic bf16 WMMA GEMM with double-buffered async global->LDS staging.
// C[M][N] = A[M][K] * Bt[N][K]^T + bias ; mode 0: fp32 out, mode 1: gelu->bf16
__global__ __launch_bounds__(256)
void gemm_bf16(const unsigned short* __restrict__ A,
               const unsigned short* __restrict__ Bt,
               const float* __restrict__ bias,
               float* __restrict__ outF, unsigned short* __restrict__ outH,
               int M, int N, int K, int mode) {
    __shared__ unsigned short As[2][128*32];
    __shared__ unsigned short Bs[2][128*32];
    int tid = threadIdx.x, lane = tid & 31, w = tid >> 5;
    int m0 = blockIdx.y * 128, n0 = blockIdx.x * 128;
    int wm = w >> 1, wn = w & 1;           // 4 x 2 wave grid, wave tile 32x64
    v8f acc[2][4] = {};

    // prologue: stage 0 into buffer 0 (4 async instructions per wave)
    #pragma unroll
    for (int i = tid; i < 512; i += 256) {
        int row = i >> 2, c8 = (i & 3) * 8;
        async_ld_b128(&As[0][row*32 + c8], &A[(size_t)(m0+row)*K + c8]);
        async_ld_b128(&Bs[0][row*32 + c8], &Bt[(size_t)(n0+row)*K + c8]);
    }

    int p = 0;
    for (int k0 = 0; k0 < K; k0 += 32, p ^= 1) {
        if (k0 + 32 < K) {
            int kn2 = k0 + 32;
            #pragma unroll
            for (int i = tid; i < 512; i += 256) {
                int row = i >> 2, c8 = (i & 3) * 8;
                async_ld_b128(&As[p^1][row*32 + c8], &A[(size_t)(m0+row)*K + kn2 + c8]);
                async_ld_b128(&Bs[p^1][row*32 + c8], &Bt[(size_t)(n0+row)*K + kn2 + c8]);
            }
            WAIT_ASYNC(4);          // retire current stage (next stage still in flight)
        } else {
            WAIT_ASYNC(0);
        }
        __syncthreads();            // current-stage tiles visible to all waves
        v16bf a0 = ldsfrag(As[p], wm*32,      32, 0, lane);
        v16bf a1 = ldsfrag(As[p], wm*32 + 16, 32, 0, lane);
        #pragma unroll
        for (int f = 0; f < 4; ++f) {
            v16bf b = ldsfrag(Bs[p], wn*64 + f*16, 32, 0, lane);
            acc[0][f] = wmma_bf16(a0, b, acc[0][f]);
            acc[1][f] = wmma_bf16(a1, b, acc[1][f]);
        }
        __syncthreads();            // all reads of buf[p] done before it is refilled
    }

    int half = lane >> 4, nl = lane & 15;
    #pragma unroll
    for (int mi = 0; mi < 2; ++mi)
        #pragma unroll
        for (int f = 0; f < 4; ++f) {
            int col = n0 + wn*64 + f*16 + nl;
            float bb = bias[col];
            #pragma unroll
            for (int r = 0; r < 8; ++r) {
                int row = m0 + wm*32 + mi*16 + r + 8*half;
                float v = acc[mi][f][r] + bb;
                if (mode == 0) outF[(size_t)row*N + col] = v;
                else           outH[(size_t)row*N + col] = f2bf(gelu_tanh(v));
            }
        }
}

// RoPE + per-head expmap0 for q,k; transpose v. One 64-thread block per (be,h,pos).
__global__ __launch_bounds__(64)
void rope_kernel(const float* __restrict__ qkv, const float* __restrict__ c_logits,
                 unsigned short* __restrict__ qh, unsigned short* __restrict__ kh,
                 unsigned short* __restrict__ vt,
                 float* __restrict__ qn2, float* __restrict__ kn2) {
    __shared__ float red[2];
    int bid = blockIdx.x;
    int pos = bid & 127, h = (bid >> 7) & 15, be = bid >> 11;
    int d = threadIdx.x, j = d & 31;
    int tok = be*128 + pos;
    const float* row = qkv + (size_t)tok * 3072;
    float freq = powf(10000.f, -(float)(2*j) / 64.f);
    float th = (float)pos * freq;
    float cs = cosf(th), sn = sinf(th);
    float qr = row[h*64 + j],        qi = row[h*64 + j + 32];
    float kr = row[1024 + h*64 + j], ki = row[1024 + h*64 + j + 32];
    float qv = (d < 32) ? qr*cs - qi*sn : qr*sn + qi*cs;
    float kv = (d < 32) ? kr*cs - ki*sn : kr*sn + ki*cs;
    float vv = row[2048 + h*64 + d];
    float c = log1pf(expf(c_logits[h]));          // softplus
    float sc = fmaxf(sqrtf(c), EPS);
    size_t rb = (size_t)(be*16 + h) * 128 + pos;
    {
        float n = sqrtf(blk64_sum(qv*qv, red));
        float safe = fmaxf(n, EPS);
        float mag = tanhf(sc*safe)/sc;
        float pn = (n < EPS) ? 0.f : mag;
        float ps = (pn >= 1.f) ? ONE_EPS/pn : 1.f;
        float coef = (n < EPS) ? 0.f : (mag/safe)*ps;
        qh[rb*64 + d] = f2bf(coef * qv);
        float pnp = (pn >= 1.f) ? ONE_EPS : pn;
        if (d == 0) qn2[rb] = pnp * pnp;
    }
    {
        float n = sqrtf(blk64_sum(kv*kv, red));
        float safe = fmaxf(n, EPS);
        float mag = tanhf(sc*safe)/sc;
        float pn = (n < EPS) ? 0.f : mag;
        float ps = (pn >= 1.f) ? ONE_EPS/pn : 1.f;
        float coef = (n < EPS) ? 0.f : (mag/safe)*ps;
        kh[rb*64 + d] = f2bf(coef * kv);
        float pnp = (pn >= 1.f) ? ONE_EPS : pn;
        if (d == 0) kn2[rb] = pnp * pnp;
    }
    vt[((size_t)(be*16 + h)*64 + d)*128 + pos] = f2bf(vv);
}

// Fused hyperbolic attention per (be,h): S = Q K^T (WMMA), closed-form mobius
// distance from dot products, softmax, O = P V (WMMA).
__global__ __launch_bounds__(256)
void attn_kernel(const unsigned short* __restrict__ qhg,
                 const unsigned short* __restrict__ khg,
                 const unsigned short* __restrict__ vtg,
                 const float* __restrict__ qn2g, const float* __restrict__ kn2g,
                 const float* __restrict__ c_logits,
                 const float* __restrict__ geo_scale,
                 unsigned short* __restrict__ ao) {
    __shared__ unsigned short QK[16384];   // Q[0..8191] K[8192..16383]; aliased as P later
    __shared__ unsigned short Vt[8192];    // [hd=64][bs=128]
    __shared__ float qn[128], kn[128];
    int tid = threadIdx.x, lane = tid & 31, w = tid >> 5;
    int bh = blockIdx.x, h = bh & 15, be = bh >> 4;
    size_t base = (size_t)bh * 8192;
    // async-stage Q, K, Vt tiles into LDS
    #pragma unroll
    for (int i = tid; i < 1024; i += 256) {
        async_ld_b128(&QK[i*8],        qhg + base + i*8);
        async_ld_b128(&QK[8192 + i*8], khg + base + i*8);
        async_ld_b128(&Vt[i*8],        vtg + base + i*8);
    }
    if (tid < 128) { qn[tid] = qn2g[(size_t)bh*128 + tid]; kn[tid] = kn2g[(size_t)bh*128 + tid]; }
    float c  = log1pf(expf(c_logits[h]));
    float sc = fmaxf(sqrtf(c), EPS);
    float gs = geo_scale[h];
    WAIT_ASYNC(0);
    __syncthreads();

    // S = Q K^T : wave w owns query rows [16w, 16w+16)
    v8f S[8] = {};
    #pragma unroll
    for (int kk = 0; kk < 64; kk += 32) {
        v16bf aq = ldsfrag(QK, w*16, 64, kk, lane);
        #pragma unroll
        for (int f = 0; f < 8; ++f) {
            v16bf bk = ldsfrag(QK + 8192, f*16, 64, kk, lane);
            S[f] = wmma_bf16(aq, bk, S[f]);
        }
    }

    // hyperbolic distance logits
    int half = lane >> 4, nl = lane & 15;
    #pragma unroll
    for (int f = 0; f < 8; ++f) {
        float yy = kn[f*16 + nl];
        #pragma unroll
        for (int r = 0; r < 8; ++r) {
            float xx = qn[w*16 + r + 8*half];
            float xy = S[f][r];
            float A  = 1.f - 2.f*c*xy + c*yy;
            float Bq = 1.f - c*xx;
            float n2 = fmaxf(A*A*xx + Bq*Bq*yy - 2.f*A*Bq*xy, 0.f);
            float den = fmaxf(1.f - 2.f*c*xy + c*c*xx*yy, EPS);
            float nr = sqrtf(n2) / den;
            nr = (nr >= 1.f) ? ONE_EPS : nr;
            float arg = fminf(sc * nr, ONE_EPS);
            S[f][r] = -gs * (2.f * atanhf(arg) / sc);
        }
    }

    // softmax over 128 keys: in-lane over 8 frags + shfl_xor over 16-lane group
    #pragma unroll
    for (int r = 0; r < 8; ++r) {
        float m = -1e30f;
        #pragma unroll
        for (int f = 0; f < 8; ++f) m = fmaxf(m, S[f][r]);
        #pragma unroll
        for (int msk = 8; msk >= 1; msk >>= 1) m = fmaxf(m, __shfl_xor(m, msk, 32));
        float s = 0.f;
        #pragma unroll
        for (int f = 0; f < 8; ++f) { S[f][r] = expf(S[f][r] - m); s += S[f][r]; }
        #pragma unroll
        for (int msk = 8; msk >= 1; msk >>= 1) s += __shfl_xor(s, msk, 32);
        float inv = 1.f / s;
        #pragma unroll
        for (int f = 0; f < 8; ++f) S[f][r] *= inv;
    }
    __syncthreads();                 // done reading Q/K tiles
    unsigned short* Ps = QK;         // alias 128x128 bf16 probs over Q/K
    #pragma unroll
    for (int f = 0; f < 8; ++f)
        #pragma unroll
        for (int r = 0; r < 8; ++r)
            Ps[(w*16 + r + 8*half)*128 + f*16 + nl] = f2bf(S[f][r]);
    __syncthreads();

    // O = P V : wave rows [16w,16w+16) x 64 cols
    v8f O[4] = {};
    #pragma unroll
    for (int kk = 0; kk < 128; kk += 32) {
        v16bf ap = ldsfrag(Ps, w*16, 128, kk, lane);
        #pragma unroll
        for (int f = 0; f < 4; ++f) {
            v16bf bv = ldsfrag(Vt, f*16, 128, kk, lane);
            O[f] = wmma_bf16(ap, bv, O[f]);
        }
    }
    #pragma unroll
    for (int f = 0; f < 4; ++f)
        #pragma unroll
        for (int r = 0; r < 8; ++r) {
            int tok = be*128 + w*16 + r + 8*half;
            int col = h*64 + f*16 + nl;
            ao[(size_t)tok*1024 + col] = f2bf(O[f][r]);
        }
}

// x_att = mobius_add(x, expmap0(ao_proj)); then prenorm2 -> t (bf16)
__global__ __launch_bounds__(256)
void postattn_kernel(const float* __restrict__ x, const float* __restrict__ aop,
                     const float* __restrict__ cg,
                     const float* __restrict__ ln2s, const float* __restrict__ ln2b,
                     float* __restrict__ xatt, unsigned short* __restrict__ tout) {
    __shared__ float red[8];
    int t = blockIdx.x, tid = threadIdx.x;
    float c = cg[0], sc = fmaxf(sqrtf(c), EPS);
    float xv[4], av[4];
    #pragma unroll
    for (int j = 0; j < 4; ++j) {
        int i = tid + j*256;
        xv[j] = x[(size_t)t*1024 + i];
        av[j] = aop[(size_t)t*1024 + i];
    }
    float ev[4]; expmap_row(av, c, sc, ev, red);
    float mv[4]; mobius_row(xv, ev, c, mv, red);
    #pragma unroll
    for (int j = 0; j < 4; ++j) xatt[(size_t)t*1024 + tid + j*256] = mv[j];
    prenorm_row(mv, c, sc, ln2s, ln2b, tout + (size_t)t*1024, red);
}

// out = mobius_add(x_att, expmap0(ffn_out))
__global__ __launch_bounds__(256)
void final_kernel(const float* __restrict__ xatt, const float* __restrict__ tt,
                  const float* __restrict__ cg, float* __restrict__ out) {
    __shared__ float red[8];
    int t = blockIdx.x, tid = threadIdx.x;
    float c = cg[0], sc = fmaxf(sqrtf(c), EPS);
    float xv[4], fv[4];
    #pragma unroll
    for (int j = 0; j < 4; ++j) {
        int i = tid + j*256;
        xv[j] = xatt[(size_t)t*1024 + i];
        fv[j] = tt[(size_t)t*1024 + i];
    }
    float ev[4]; expmap_row(fv, c, sc, ev, red);
    float mv[4]; mobius_row(xv, ev, c, mv, red);
    #pragma unroll
    for (int j = 0; j < 4; ++j) out[(size_t)t*1024 + tid + j*256] = mv[j];
}

// ---------- launch ----------

extern "C" void kernel_launch(void* const* d_in, const int* in_sizes, int n_in,
                              void* d_out, int out_size, void* d_ws, size_t ws_size,
                              hipStream_t stream) {
    const float* x         = (const float*)d_in[0];
    const float* c_global  = (const float*)d_in[1];
    const float* qkv_w     = (const float*)d_in[2];
    const float* qkv_b     = (const float*)d_in[3];
    const float* out_w     = (const float*)d_in[4];
    const float* out_b     = (const float*)d_in[5];
    const float* ffn_w1    = (const float*)d_in[6];
    const float* ffn_b1    = (const float*)d_in[7];
    const float* ffn_w2    = (const float*)d_in[8];
    const float* ffn_b2    = (const float*)d_in[9];
    const float* ln1_scale = (const float*)d_in[10];
    const float* ln1_bias  = (const float*)d_in[11];
    const float* ln2_scale = (const float*)d_in[12];
    const float* ln2_bias  = (const float*)d_in[13];
    const float* c_logits  = (const float*)d_in[14];
    const float* geo_scale = (const float*)d_in[15];
    float* outp = (float*)d_out;

    char* ws = (char*)d_ws;
    auto alloc = [&](size_t bytes) -> char* {
        char* p = ws; ws += (bytes + 255) & ~(size_t)255; return p;
    };
    unsigned short* qkv_wT = (unsigned short*)alloc((size_t)3072*1024*2);
    unsigned short* out_wT = (unsigned short*)alloc((size_t)1024*1024*2);
    unsigned short* w1T    = (unsigned short*)alloc((size_t)4096*1024*2);
    unsigned short* w2T    = (unsigned short*)alloc((size_t)1024*4096*2);
    unsigned short* xtan   = (unsigned short*)alloc((size_t)1024*1024*2);
    float*          qkvF   = (float*)         alloc((size_t)1024*3072*4);
    unsigned short* qh     = (unsigned short*)alloc((size_t)1024*1024*2);
    unsigned short* kh     = (unsigned short*)alloc((size_t)1024*1024*2);
    unsigned short* vt     = (unsigned short*)alloc((size_t)1024*1024*2);
    float*          qn2    = (float*)         alloc((size_t)16384*4);
    float*          kn2    = (float*)         alloc((size_t)16384*4);
    unsigned short* ao     = (unsigned short*)alloc((size_t)1024*1024*2);
    float*          aoproj = (float*)         alloc((size_t)1024*1024*4);
    float*          xatt   = (float*)         alloc((size_t)1024*1024*4);
    unsigned short* tbuf   = (unsigned short*)alloc((size_t)1024*1024*2);
    unsigned short* hbuf   = (unsigned short*)alloc((size_t)1024*4096*2);
    float*          ttbuf  = (float*)         alloc((size_t)1024*1024*4);

    // weight conversion (fp32 [K][N] -> bf16 [N][K])
    wconv_kernel<<<4096, 256, 0, stream>>>(qkv_w,  qkv_wT, 1024, 3072);
    wconv_kernel<<<4096, 256, 0, stream>>>(out_w,  out_wT, 1024, 1024);
    wconv_kernel<<<4096, 256, 0, stream>>>(ffn_w1, w1T,    1024, 4096);
    wconv_kernel<<<4096, 256, 0, stream>>>(ffn_w2, w2T,    4096, 1024);

    // pre-norm 1 -> x_tan (bf16)
    prenorm1_kernel<<<1024, 256, 0, stream>>>(x, c_global, ln1_scale, ln1_bias, xtan);

    // QKV projection
    gemm_bf16<<<dim3(24, 8), 256, 0, stream>>>(xtan, qkv_wT, qkv_b,
                                               qkvF, nullptr, 1024, 3072, 1024, 0);

    // RoPE + head-wise expmap0
    rope_kernel<<<16384, 64, 0, stream>>>(qkvF, c_logits, qh, kh, vt, qn2, kn2);

    // fused hyperbolic attention
    attn_kernel<<<128, 256, 0, stream>>>(qh, kh, vt, qn2, kn2, c_logits, geo_scale, ao);

    // output projection
    gemm_bf16<<<dim3(8, 8), 256, 0, stream>>>(ao, out_wT, out_b,
                                              aoproj, nullptr, 1024, 1024, 1024, 0);

    // residual mobius add + pre-norm 2 -> t (bf16)
    postattn_kernel<<<1024, 256, 0, stream>>>(x, aoproj, c_global,
                                              ln2_scale, ln2_bias, xatt, tbuf);

    // FFN
    gemm_bf16<<<dim3(32, 8), 256, 0, stream>>>(tbuf, w1T, ffn_b1,
                                               nullptr, hbuf, 1024, 4096, 1024, 1);
    gemm_bf16<<<dim3(8, 8), 256, 0, stream>>>(hbuf, w2T, ffn_b2,
                                              ttbuf, nullptr, 1024, 1024, 4096, 0);

    // final expmap + mobius residual
    final_kernel<<<1024, 256, 0, stream>>>(xatt, ttbuf, c_global, outp);
}
